// Convnet_14310831031028
// MI455X (gfx1250) — matrix operand order repaired and checked by reading, not compiled, not run on hardware
//
#include <hip/hip_runtime.h>
#include <stdint.h>

// ---------------------------------------------------------------- constants
#define NCH   64          // NC
#define DEF   32          // DE
#define HDIM  128         // H
#define NNODE 10000
#define NEDGE 200000
#define INV_DIV (1.0f/20.0f)

// packed-weight offsets in workspace (uint32 units)
#define OFF_R1W1 0
#define OFF_R1W2 4096
#define OFF_R2W1 20480
#define OFF_R2W2 24576
#define OFF_L1WS 40960
#define OFF_L1WV 49152
#define OFF_L2WS 53248
#define OFF_L2WV 61440
#define OFF_L3WS 65536
#define OFF_L3WV 67584
#define OFF_SWS  69632
#define OFF_SWV  71680
#define OFF_SA   73728                  // float index: sa accumulator (N x 64)
#define OFF_VA   (OFF_SA + NNODE*NCH)   // va accumulator (N x 64 x 3)
#define ACC_COUNT (NNODE*NCH*4)

// per-wave LDS layout (bytes / half-strides)
enum { WSTR = 264, HSTR = 136, SESTR = 72, VSTR = 72, PLANE = 16*VSTR };
enum { WBYTES = 16*WSTR*2, HBYTES = 16*HSTR*2, SEBYTES = 16*SESTR*2,
       VMBYTES = 3*16*VSTR*2, PER_WAVE = WBYTES+HBYTES+SEBYTES+VMBYTES };
#define SMEM_EDGE (2*PER_WAVE)

typedef _Float16 half16 __attribute__((ext_vector_type(16)));
typedef float    float8 __attribute__((ext_vector_type(8)));

union AFrag { half16 h; uint32_t u[8]; };
union BFrag { half16 h; uint4 q[2]; };
union HPair { uint32_t u; _Float16 h[2]; };

__device__ __forceinline__ float8 wmma_f16(half16 a, half16 b, float8 c) {
  return __builtin_amdgcn_wmma_f32_16x16x32_f16(false, a, false, b, (short)0, c,
                                                false, false);
}

// fast sigmoid: v_exp_f32 + v_add + v_rcp (hardware reciprocal, no IEEE div chain)
__device__ __forceinline__ float sigm(float x) {
  return __builtin_amdgcn_rcpf(1.0f + __expf(-x));
}

// B fragment: packed[((chunk*ntiles + t)*32 + lane)*8 + j]
__device__ __forceinline__ half16 load_bfrag(const uint32_t* packed, int chunk,
                                             int ntiles, int t, int lane) {
  const uint4* p = (const uint4*)(packed + (size_t)((chunk * ntiles + t) * 32 + lane) * 8);
  BFrag b;
  b.q[0] = p[0];
  b.q[1] = p[1];
  return b.h;
}

// A fragment K-offset within a 32-wide chunk for register j (16-bit A layout)
__device__ __forceinline__ int a_koff(int j, int kbA) {
  return ((j >> 2) << 4) + kbA + ((j & 3) << 1);
}

// A fragment from an LDS row (f16, row-major, even stride)
__device__ __forceinline__ half16 load_afrag_lds(const _Float16* rowp, int kbase, int kbA) {
  AFrag a;
#pragma unroll
  for (int j = 0; j < 8; ++j) {
    int k = kbase + a_koff(j, kbA);
    a.u[j] = *(const uint32_t*)(rowp + k);
  }
  return a.h;
}

// A fragment from global f32 with element stride (converts to f16)
__device__ __forceinline__ half16 make_afrag_global(const float* p, int kstride,
                                                    int kbase, int kbA) {
  AFrag a;
#pragma unroll
  for (int j = 0; j < 8; ++j) {
    int k = kbase + a_koff(j, kbA);
    HPair pv;
    pv.h[0] = (_Float16)p[(size_t)k * kstride];
    pv.h[1] = (_Float16)p[(size_t)(k + 1) * kstride];
    a.u[j] = pv.u;
  }
  return a.h;
}

// ------------------------------------------------------------- radial MLP
// feat(16x33 pad 64) @ W1(64x128) -> silu -> @ W2(128x256) -> wbuf (f16)
__device__ __forceinline__ void radial_stage(int e0, int lane, const float* r_ij,
                                             const float* edges, const uint32_t* W1p,
                                             const float* b1, const uint32_t* W2p,
                                             const float* b2, _Float16* hbuf,
                                             _Float16* wbuf) {
  const int rowA = lane & 15, hi = lane >> 4, kbA = hi * 8, colC = lane & 15;
  half16 af[2];
#pragma unroll
  for (int c = 0; c < 2; ++c) {
    AFrag a;
#pragma unroll
    for (int j = 0; j < 8; ++j) {
      int k = c * 32 + a_koff(j, kbA);
      float f0 = (k == 0) ? r_ij[e0 + rowA]
                          : ((k < 33) ? edges[(size_t)(e0 + rowA) * DEF + (k - 1)] : 0.f);
      float f1 = ((k + 1) < 33) ? edges[(size_t)(e0 + rowA) * DEF + k] : 0.f;
      HPair pv; pv.h[0] = (_Float16)f0; pv.h[1] = (_Float16)f1;
      a.u[j] = pv.u;
    }
    af[c] = a.h;
  }
#pragma unroll
  for (int t = 0; t < 8; ++t) {
    float bv = b1[t * 16 + colC];
    float8 acc;
#pragma unroll
    for (int r = 0; r < 8; ++r) acc[r] = bv;
    acc = wmma_f16(af[0], load_bfrag(W1p, 0, 8, t, lane), acc);
    acc = wmma_f16(af[1], load_bfrag(W1p, 1, 8, t, lane), acc);
#pragma unroll
    for (int r = 0; r < 8; ++r) {
      float x = acc[r];
      hbuf[(r + 8 * hi) * HSTR + t * 16 + colC] = (_Float16)(x * sigm(x));
    }
  }
  asm volatile("s_wait_dscnt 0x0" ::: "memory");
  half16 ah[4];
  const _Float16* hrow = hbuf + rowA * HSTR;
#pragma unroll
  for (int c = 0; c < 4; ++c) ah[c] = load_afrag_lds(hrow, c * 32, kbA);
#pragma unroll
  for (int t = 0; t < 16; ++t) {
    float bv = b2[t * 16 + colC];
    float8 acc;
#pragma unroll
    for (int r = 0; r < 8; ++r) acc[r] = bv;
#pragma unroll
    for (int c = 0; c < 4; ++c)
      acc = wmma_f16(ah[c], load_bfrag(W2p, c, 16, t, lane), acc);
#pragma unroll
    for (int r = 0; r < 8; ++r)
      wbuf[(r + 8 * hi) * WSTR + t * 16 + colC] = (_Float16)acc[r];
  }
  asm volatile("s_wait_dscnt 0x0" ::: "memory");
}

// ------------------------------------------------------------- edge kernel
__global__ void edge_kernel(const float* __restrict__ s, const float* __restrict__ v,
                            const float* __restrict__ edges, const float* __restrict__ r_ij,
                            const float* __restrict__ rvec, const float* __restrict__ rb11,
                            const float* __restrict__ rb12, const float* __restrict__ rb21,
                            const float* __restrict__ rb22, const float* __restrict__ l1bs,
                            const float* __restrict__ l2bs, const int* __restrict__ srcI,
                            const int* __restrict__ dstI, const uint32_t* __restrict__ wpk,
                            float* __restrict__ sa, float* __restrict__ va) {
  extern __shared__ char smem[];
  const int wave = threadIdx.x >> 5, lane = threadIdx.x & 31;
  const int tile = blockIdx.x * (blockDim.x >> 5) + wave;
  if (tile >= NEDGE / 16) return;
  const int e0 = tile * 16;

  char* base = smem + (size_t)wave * PER_WAVE;
  _Float16* wbuf  = (_Float16*)base;
  _Float16* hbuf  = (_Float16*)(base + WBYTES);
  _Float16* sebuf = (_Float16*)(base + WBYTES + HBYTES);
  _Float16* vmbuf = (_Float16*)(base + WBYTES + HBYTES + SEBYTES);

  const int rowA = lane & 15, hi = lane >> 4, kbA = hi * 8, colC = lane & 15;
  const int di = dstI[e0 + rowA], si = srcI[e0 + rowA];
  const float* sdp = s + (size_t)di * NCH;
  const float* ssp = s + (size_t)si * NCH;
  const float* vdp = v + (size_t)di * NCH * 3;
  const float* vsp = v + (size_t)si * NCH * 3;
  const float rv0 = rvec[(size_t)(e0 + rowA) * 3 + 0];
  const float rv1 = rvec[(size_t)(e0 + rowA) * 3 + 1];
  const float rv2 = rvec[(size_t)(e0 + rowA) * 3 + 2];
  const _Float16* wrow = wbuf + rowA * WSTR;

  auto mkA = [&](int kbase, auto f) -> half16 {
    AFrag a;
#pragma unroll
    for (int j = 0; j < 8; ++j) {
      int k = kbase + a_koff(j, kbA);
      HPair pv;
      pv.h[0] = (_Float16)f(k);
      pv.h[1] = (_Float16)f(k + 1);
      a.u[j] = pv.u;
    }
    return a.h;
  };
  auto smval = [&](int k) -> float {
    if (k < NCH) return sdp[k] * ssp[k];
    int kk = k - NCH;
    return vdp[kk * 3] * vsp[kk * 3] + vdp[kk * 3 + 1] * vsp[kk * 3 + 1] +
           vdp[kk * 3 + 2] * vsp[kk * 3 + 2];
  };

  // ---- radial 1 -> wbuf (w1, 16x256 f16)
  radial_stage(e0, lane, r_ij, edges, wpk + OFF_R1W1, rb11, wpk + OFF_R1W2, rb12,
               hbuf, wbuf);

  // ---- lin1 scalar path: A = w1[:, :128] * sm
  half16 as4[4];
#pragma unroll
  for (int c = 0; c < 4; ++c)
    as4[c] = mkA(c * 32, [&](int k) { return smval(k) * (float)wrow[k]; });
  float8 sm1[8];
#pragma unroll
  for (int t = 0; t < 8; ++t) {
    float bv = l1bs[t * 16 + colC];
    float8 acc;
#pragma unroll
    for (int r = 0; r < 8; ++r) acc[r] = bv;
#pragma unroll
    for (int c = 0; c < 4; ++c)
      acc = wmma_f16(as4[c], load_bfrag(wpk + OFF_L1WS, c, 8, t, lane), acc);
    sm1[t] = acc;
  }
  float8 seF[4], gF[4];
#pragma unroll
  for (int t = 0; t < 4; ++t) {
#pragma unroll
    for (int r = 0; r < 8; ++r) {
      float x = sm1[t][r];
      seF[t][r] = x * sigm(x);
      gF[t][r] = sigm(sm1[4 + t][r]);
    }
  }
#pragma unroll
  for (int t = 0; t < 4; ++t)
#pragma unroll
    for (int r = 0; r < 8; ++r)
      sebuf[(r + 8 * hi) * SESTR + t * 16 + colC] = (_Float16)seF[t][r];

  // ---- lin1 vector path per component: A = w1[:, 128:] * vm, gated, to LDS
#pragma unroll
  for (int cc = 0; cc < 3; ++cc) {
    half16 av[4];
#pragma unroll
    for (int c = 0; c < 4; ++c)
      av[c] = mkA(c * 32, [&](int k) {
        float b = (k < NCH) ? sdp[k] * vsp[k * 3 + cc]
                            : vdp[(k - NCH) * 3 + cc] * ssp[k - NCH];
        return b * (float)wrow[HDIM + k];
      });
#pragma unroll
    for (int t = 0; t < 4; ++t) {
      float8 acc;
#pragma unroll
      for (int r = 0; r < 8; ++r) acc[r] = 0.f;
#pragma unroll
      for (int c = 0; c < 4; ++c)
        acc = wmma_f16(av[c], load_bfrag(wpk + OFF_L1WV, c, 4, t, lane), acc);
#pragma unroll
      for (int r = 0; r < 8; ++r)
        vmbuf[cc * PLANE + (r + 8 * hi) * VSTR + t * 16 + colC] =
            (_Float16)(acc[r] * gF[t][r]);
    }
  }
  asm volatile("s_wait_dscnt 0x0" ::: "memory");

  // ---- radial 2 -> wbuf (reused)
  radial_stage(e0, lane, r_ij, edges, wpk + OFF_R2W1, rb21, wpk + OFF_R2W2, rb22,
               hbuf, wbuf);

  // ---- lin2 scalar path: A = w2[:, :128] * concat(se, sum_c vm*rvec)
  half16 as2[4];
#pragma unroll
  for (int c = 0; c < 4; ++c)
    as2[c] = mkA(c * 32, [&](int k) {
      float b;
      if (k < NCH) {
        b = (float)sebuf[rowA * SESTR + k];
      } else {
        int kk = k - NCH;
        b = (float)vmbuf[0 * PLANE + rowA * VSTR + kk] * rv0 +
            (float)vmbuf[1 * PLANE + rowA * VSTR + kk] * rv1 +
            (float)vmbuf[2 * PLANE + rowA * VSTR + kk] * rv2;
      }
      return b * (float)wrow[k];
    });
  float8 sm3[8];
#pragma unroll
  for (int t = 0; t < 8; ++t) {
    float bv = l2bs[t * 16 + colC];
    float8 acc;
#pragma unroll
    for (int r = 0; r < 8; ++r) acc[r] = bv;
#pragma unroll
    for (int c = 0; c < 4; ++c)
      acc = wmma_f16(as2[c], load_bfrag(wpk + OFF_L2WS, c, 8, t, lane), acc);
    sm3[t] = acc;
  }
  float8 se2[4], g2[4];
#pragma unroll
  for (int t = 0; t < 4; ++t) {
#pragma unroll
    for (int r = 0; r < 8; ++r) {
      float x = sm3[t][r];
      se2[t][r] = x * sigm(x);
      g2[t][r] = sigm(sm3[4 + t][r]);
    }
  }

  int drow[8];
#pragma unroll
  for (int r = 0; r < 8; ++r) drow[r] = dstI[e0 + r + 8 * hi];

  // scalar scatter-add (with 1/DIV folded in)
#pragma unroll
  for (int t = 0; t < 4; ++t) {
    int col = t * 16 + colC;
#pragma unroll
    for (int r = 0; r < 8; ++r)
      atomicAdd(&sa[(size_t)drow[r] * NCH + col], se2[t][r] * INV_DIV);
  }

  // ---- lin2 vector path + vector scatter-add
  const float rvs[3] = {rv0, rv1, rv2};
#pragma unroll
  for (int cc = 0; cc < 3; ++cc) {
    half16 av[4];
#pragma unroll
    for (int c = 0; c < 4; ++c)
      av[c] = mkA(c * 32, [&](int k) {
        float b = (k < NCH) ? (float)sebuf[rowA * SESTR + k] * rvs[cc]
                            : (float)vmbuf[cc * PLANE + rowA * VSTR + (k - NCH)];
        return b * (float)wrow[HDIM + k];
      });
#pragma unroll
    for (int t = 0; t < 4; ++t) {
      float8 acc;
#pragma unroll
      for (int r = 0; r < 8; ++r) acc[r] = 0.f;
#pragma unroll
      for (int c = 0; c < 4; ++c)
        acc = wmma_f16(av[c], load_bfrag(wpk + OFF_L2WV, c, 4, t, lane), acc);
      int col = t * 16 + colC;
#pragma unroll
      for (int r = 0; r < 8; ++r)
        atomicAdd(&va[((size_t)drow[r] * NCH + col) * 3 + cc],
                  acc[r] * g2[t][r] * INV_DIV);
    }
  }
}

// ------------------------------------------------------------- node kernel
__global__ void node_kernel(const float* __restrict__ s, const float* __restrict__ v,
                            const float* __restrict__ l3bs,
                            const uint32_t* __restrict__ wpk,
                            const float* __restrict__ sa, const float* __restrict__ va,
                            float* __restrict__ out) {
  const int wave = threadIdx.x >> 5, lane = threadIdx.x & 31;
  const int tile = blockIdx.x * (blockDim.x >> 5) + wave;
  if (tile >= NNODE / 16) return;
  const int n0 = tile * 16;
  const int rowA = lane & 15, hi = lane >> 4, kbA = hi * 8, colC = lane & 15;

  const float* saRow = sa + (size_t)(n0 + rowA) * NCH;
  const float* sRow  = s  + (size_t)(n0 + rowA) * NCH;
  const float* vaRow = va + (size_t)(n0 + rowA) * NCH * 3;
  const float* vRow  = v  + (size_t)(n0 + rowA) * NCH * 3;

  half16 aSA[2], aS[2];
#pragma unroll
  for (int c = 0; c < 2; ++c) {
    aSA[c] = make_afrag_global(saRow, 1, c * 32, kbA);
    aS[c]  = make_afrag_global(sRow, 1, c * 32, kbA);
  }
#pragma unroll
  for (int t = 0; t < 4; ++t) {
    float bv = l3bs[t * 16 + colC];
    float8 acc;
#pragma unroll
    for (int r = 0; r < 8; ++r) acc[r] = bv;
#pragma unroll
    for (int c = 0; c < 2; ++c) {
      acc = wmma_f16(aSA[c], load_bfrag(wpk + OFF_L3WS, c, 4, t, lane), acc);
      acc = wmma_f16(aS[c],  load_bfrag(wpk + OFF_SWS,  c, 4, t, lane), acc);
    }
#pragma unroll
    for (int r = 0; r < 8; ++r)
      out[(size_t)(n0 + r + 8 * hi) * NCH + t * 16 + colC] = acc[r];
  }
#pragma unroll
  for (int cc = 0; cc < 3; ++cc) {
    half16 aVA[2], aV[2];
#pragma unroll
    for (int c = 0; c < 2; ++c) {
      aVA[c] = make_afrag_global(vaRow + cc, 3, c * 32, kbA);
      aV[c]  = make_afrag_global(vRow + cc, 3, c * 32, kbA);
    }
#pragma unroll
    for (int t = 0; t < 4; ++t) {
      float8 acc;
#pragma unroll
      for (int r = 0; r < 8; ++r) acc[r] = 0.f;
#pragma unroll
      for (int c = 0; c < 2; ++c) {
        acc = wmma_f16(aVA[c], load_bfrag(wpk + OFF_L3WV, c, 4, t, lane), acc);
        acc = wmma_f16(aV[c],  load_bfrag(wpk + OFF_SWV,  c, 4, t, lane), acc);
      }
#pragma unroll
      for (int r = 0; r < 8; ++r)
        out[(size_t)NNODE * NCH +
            ((size_t)(n0 + r + 8 * hi) * NCH + t * 16 + colC) * 3 + cc] = acc[r];
    }
  }
}

// --------------------------------------------------------- weight packing
// Pack W (K x Nn row-major f32, zero-pad to Kp) into B-fragment order f16 pairs.
__global__ void pack_kernel(const float* __restrict__ W, int K, int Kp, int Nn,
                            uint32_t* __restrict__ out) {
  int idx = blockIdx.x * blockDim.x + threadIdx.x;
  int total = (Kp >> 5) * (Nn >> 4) * 256;
  if (idx >= total) return;
  int j = idx & 7, lane = (idx >> 3) & 31, rest = idx >> 8;
  int ntiles = Nn >> 4;
  int t = rest % ntiles, chunk = rest / ntiles;
  int k = chunk * 32 + ((j >> 2) << 4) + ((lane >> 4) << 3) + ((j & 3) << 1);
  int n = t * 16 + (lane & 15);
  float w0 = (k < K) ? W[(size_t)k * Nn + n] : 0.f;
  float w1 = (k + 1 < K) ? W[(size_t)(k + 1) * Nn + n] : 0.f;
  HPair pv;
  pv.h[0] = (_Float16)w0;
  pv.h[1] = (_Float16)w1;
  out[idx] = pv.u;
}

__global__ void zero_kernel(float* __restrict__ p, int n) {
  int i = blockIdx.x * blockDim.x + threadIdx.x;
  if (i < n) p[i] = 0.f;
}

// ------------------------------------------------------------------ launch
extern "C" void kernel_launch(void* const* d_in, const int* in_sizes, int n_in,
                              void* d_out, int out_size, void* d_ws, size_t ws_size,
                              hipStream_t stream) {
  (void)in_sizes; (void)n_in; (void)out_size; (void)ws_size;
  const float* s        = (const float*)d_in[0];
  const float* v        = (const float*)d_in[1];
  const float* edges    = (const float*)d_in[2];
  const float* r_ij     = (const float*)d_in[3];
  const float* rvec     = (const float*)d_in[4];
  const float* rb11     = (const float*)d_in[6];
  const float* rb12     = (const float*)d_in[8];
  const float* rb21     = (const float*)d_in[10];
  const float* rb22     = (const float*)d_in[12];
  const float* l1bs     = (const float*)d_in[14];
  const float* l2bs     = (const float*)d_in[17];
  const float* l3bs     = (const float*)d_in[20];
  const int*   srcI     = (const int*)d_in[24];
  const int*   dstI     = (const int*)d_in[25];

  uint32_t* wpk = (uint32_t*)d_ws;
  float*    wsf = (float*)d_ws;
  float*    sa  = wsf + OFF_SA;
  float*    va  = wsf + OFF_VA;
  float*    out = (float*)d_out;

  auto packW = [&](int in_idx, int K, int Kp, int Nn, int off) {
    int total = (Kp / 32) * (Nn / 16) * 256;
    pack_kernel<<<(total + 255) / 256, 256, 0, stream>>>((const float*)d_in[in_idx],
                                                         K, Kp, Nn, wpk + off);
  };
  packW(5,  33,  64, 128, OFF_R1W1);
  packW(7,  128, 128, 256, OFF_R1W2);
  packW(9,  33,  64, 128, OFF_R2W1);
  packW(11, 128, 128, 256, OFF_R2W2);
  packW(13, 128, 128, 128, OFF_L1WS);
  packW(15, 128, 128, 64,  OFF_L1WV);
  packW(16, 128, 128, 128, OFF_L2WS);
  packW(18, 128, 128, 64,  OFF_L2WV);
  packW(19, 64,  64, 64,   OFF_L3WS);
  packW(21, 64,  64, 64,   OFF_L3WV);
  packW(22, 64,  64, 64,   OFF_SWS);
  packW(23, 64,  64, 64,   OFF_SWV);

  zero_kernel<<<(ACC_COUNT + 255) / 256, 256, 0, stream>>>(sa, ACC_COUNT);

  // 12500 edge tiles, 2 waves (tiles) per 64-thread block
  edge_kernel<<<NEDGE / 16 / 2, 64, SMEM_EDGE, stream>>>(
      s, v, edges, r_ij, rvec, rb11, rb12, rb21, rb22, l1bs, l2bs, srcI, dstI,
      wpk, sa, va);

  // 625 node tiles, 2 per block
  node_kernel<<<(NNODE / 16 + 1) / 2, 64, 0, stream>>>(s, v, l3bs, wpk, sa, va, out);
}
